// SemanticDecoder_58720792871457
// MI455X (gfx1250) — matrix operand order repaired and checked by reading, not compiled, not run on hardware
//
#include <hip/hip_runtime.h>

// bf16 storage = unsigned short; manual conversions (RNE) keep host pass clean.
typedef __attribute__((ext_vector_type(16))) __bf16 v16bf;
typedef __attribute__((ext_vector_type(8)))  float  v8f;

// TDM descriptor SGPR-group types (clang-23 / amdgpu-toolchain 6-arg builtin).
typedef unsigned int u32x4 __attribute__((ext_vector_type(4)));
typedef int          i32x8 __attribute__((ext_vector_type(8)));
typedef int          i32x4 __attribute__((ext_vector_type(4)));

#define T_STEPS 4096
#define HID     512
#define GATES   2048   // 4*HID, gate order i,f,g,o

__device__ __forceinline__ unsigned short f2bf(float f) {
  unsigned int u = __float_as_uint(f);
  unsigned int r = u + 0x7FFFu + ((u >> 16) & 1u);   // round-to-nearest-even
  return (unsigned short)(r >> 16);
}
__device__ __forceinline__ float bf2f(unsigned short h) {
  return __uint_as_float(((unsigned int)h) << 16);
}

// CDNA5 async memory->LDS copy (ASYNCcnt-tracked, no VGPR staging).
// LDS destination = low 32 bits of the flat pointer (aperture uses addr[31:0]).
__device__ __forceinline__ void async_load_b128(const void* lds_dst,
                                                const void* gsrc) {
  unsigned lds_off = (unsigned)(unsigned long long)lds_dst;
  asm volatile("global_load_async_to_lds_b128 %0, %1, off"
               :: "v"(lds_off), "v"(gsrc) : "memory");
}
__device__ __forceinline__ void wait_async0() {
  asm volatile("s_wait_asynccnt 0x0" ::: "memory");
}

// ---------------- utility kernels ----------------
__global__ void cvt_f32_bf16_kernel(const float* __restrict__ s,
                                    unsigned short* __restrict__ d, int n) {
  int i = blockIdx.x * 256 + threadIdx.x;
  if (i < n) d[i] = f2bf(s[i]);
}

__global__ void add_vec_kernel(const float* __restrict__ a,
                               const float* __restrict__ b,
                               float* __restrict__ o, int n) {
  int i = blockIdx.x * 256 + threadIdx.x;
  if (i < n) o[i] = a[i] + b[i];
}

__global__ void init_barrier_kernel(int* p) {
  if (threadIdx.x == 0) { p[0] = 0; p[1] = 0; }
}

// ---------------- WMMA GEMM: out[M][N] = Xbf[M][K] @ Wbf[N][K]^T + bias[N] ----
// Block tile 128(M) x 128(N), K-step 32; 8 waves, each computing 32M x 64N
// (2 A-frags x 4 B-frags = 8 wmma per K-step). Double-buffered async->LDS
// staging: DMA for step i+1 overlaps the WMMAs of step i; one barrier/step.
#define TM  128
#define TN  128
#define TKS 32

__global__ __launch_bounds__(256)
void gemm_bias_wmma_kernel(const unsigned short* __restrict__ X,
                           const unsigned short* __restrict__ W,
                           const float* __restrict__ bias,
                           float* __restrict__ out,
                           int M, int N, int K)
{
  __shared__ __attribute__((aligned(32))) unsigned short As[2][TM * TKS]; // 16KB
  __shared__ __attribute__((aligned(32))) unsigned short Bs[2][TN * TKS]; // 16KB

  const int tid  = threadIdx.x;
  const int lane = tid & 31;
  const int wave = tid >> 5;              // 0..7
  const int m0 = blockIdx.y * TM;
  const int n0 = blockIdx.x * TN;
  const int wm = (wave & 3) * 32;         // wave's M offset in block tile
  const int wn = (wave >> 2) * 64;        // wave's N offset in block tile

  v8f acc[2][4] = {};

  // A fragment addressing (16x32 bf16): lanes 0-15 -> K {0..7,16..23},
  // lanes 16-31 -> K {8..15,24..31}; row = lane&15.
  const int ar = lane & 15;
  const int ak = (lane < 16) ? 0 : 8;
  // B fragment addressing (32x16 bf16, stored n-major k-contiguous):
  // lanes 0-15 -> K {0..7,16..23}, lanes 16-31 -> K {8..15,24..31}; col n = lane&15.
  const int bn = lane & 15;
  const int bk = (lane < 16) ? 0 : 8;

  // Issue one K-tile (A 128x32 + B 128x32 = 1024 16B chunks) into buffer `buf`.
  auto issue_tile = [&](int k0, int buf) {
    #pragma unroll
    for (int c = 0; c < 2; ++c) {
      int idx = tid + c * 256;
      int row = idx >> 2;
      int col = (idx & 3) << 3;
      async_load_b128(&As[buf][row * TKS + col],
                      X + (size_t)(m0 + row) * K + (k0 + col));
      async_load_b128(&Bs[buf][row * TKS + col],
                      W + (size_t)(n0 + row) * K + (k0 + col));
    }
  };

  const int niter = K / TKS;
  issue_tile(0, 0);

  for (int i = 0; i < niter; ++i) {
    const int cur = i & 1;
    wait_async0();        // my chunks of buffer `cur` have landed in LDS
    __syncthreads();      // everyone's chunks landed; everyone done reading cur^1
    if (i + 1 < niter) issue_tile((i + 1) * TKS, cur ^ 1);  // DMA overlaps WMMA

    union frag { v16bf v; uint4 u[2]; };
    frag a[2];
    #pragma unroll
    for (int ii = 0; ii < 2; ++ii) {
      a[ii].u[0] = *(const uint4*)(&As[cur][(wm + ii * 16 + ar) * TKS + ak]);
      a[ii].u[1] = *(const uint4*)(&As[cur][(wm + ii * 16 + ar) * TKS + ak + 16]);
    }
    #pragma unroll
    for (int j = 0; j < 4; ++j) {
      frag b;
      b.u[0] = *(const uint4*)(&Bs[cur][(wn + j * 16 + bn) * TKS + bk]);
      b.u[1] = *(const uint4*)(&Bs[cur][(wn + j * 16 + bn) * TKS + bk + 16]);
      #pragma unroll
      for (int ii = 0; ii < 2; ++ii) {
        acc[ii][j] = __builtin_amdgcn_wmma_f32_16x16x32_bf16(
                         false, a[ii].v, false, b.v, (short)0, acc[ii][j],
                         false, false);
      }
    }
  }

  // D layout: VGPR r -> M=r (lanes 0-15) / M=r+8 (lanes 16-31); N = lane&15.
  const int dn = lane & 15;
  const int dm = (lane < 16) ? 0 : 8;
  #pragma unroll
  for (int j = 0; j < 4; ++j) {
    float bv = bias[n0 + wn + j * 16 + dn];
    #pragma unroll
    for (int i = 0; i < 2; ++i) {
      #pragma unroll
      for (int r = 0; r < 8; ++r) {
        out[(size_t)(m0 + wm + i * 16 + dm + r) * N + (n0 + wn + j * 16 + dn)] =
            acc[i][j][r] + bv;
      }
    }
  }
}

// ---------------- persistent LSTM scan ----------------
// 64 blocks x 256 threads. Block b owns hidden units [8b, 8b+8): its 32 gate
// rows (i,f,g,o for those units) of W_hh stay LDS-resident (32KB), pulled in
// by one TDM 3D-tile descriptor (4 gate-groups x 8 rows x 512 bf16). Per step:
// async-broadcast h[t-1] (1KB from L2), 32 x 512 dot products, local pointwise
// update (c in registers), publish h[t], grid barrier.
__global__ __launch_bounds__(256)
void lstm_scan_kernel(const float* __restrict__ G,            // [T][2048] gates_x
                      const unsigned short* __restrict__ Whh, // [2048][512] bf16
                      unsigned short* __restrict__ Hout,      // [T][512] bf16
                      int* __restrict__ bar,                  // {count, generation}
                      int nsteps)
{
  __shared__ __attribute__((aligned(32))) unsigned short wlds[32 * HID]; // 32KB
  __shared__ __attribute__((aligned(32))) unsigned short hlds[HID];      // 1KB
  __shared__ float red[32][8];

  const int tid = threadIdx.x;
  const int blk = blockIdx.x;       // 0..63
  const int j0  = blk * 8;          // hidden-unit slice

#if __has_builtin(__builtin_amdgcn_tensor_load_to_lds)
  // TDM load: tile_dim0=512 (row), tile_dim1=8 (units), tile_dim2=4 (gates);
  // tensor dim0=512 stride 512, dim1=512 rows stride 512*512, dim2=4.
  // Local row lr = gate*8 + unit lands at wlds + lr*512 (LDS-contiguous tile).
  if (tid < 32) {                   // one wave issues the DMA; EXEC is ignored
    unsigned long long ga = (unsigned long long)(const void*)(Whh + (size_t)j0 * HID);
    unsigned ldsb = (unsigned)(unsigned long long)(const void*)wlds;
    u32x4 g0;
    g0[0] = 1u;                                   // count=1, user descriptor
    g0[1] = ldsb;                                 // lds_addr (bytes)
    g0[2] = (unsigned)ga;                         // global_addr[31:0]
    g0[3] = (unsigned)(ga >> 32) | (2u << 30);    // global_addr[56:32] | type=2
    i32x8 g1;
    g1[0] = (int)(1u << 16);                      // data_size=1 (2 bytes/elt)
    g1[1] = (int)(512u << 16);                    // tensor_dim0 = 512
    g1[2] = (int)(512u << 16);                    // tensor_dim1 = 512
    g1[3] = (int)(512u << 16);                    // tile_dim0 = 512
    g1[4] = (int)(8u | (4u << 16));               // tile_dim1=8, tile_dim2=4
    g1[5] = 512;                                  // tensor_dim0_stride lo32
    g1[6] = 0;                                    // strides mid bits
    g1[7] = 4;                                    // tensor_dim1_stride = 4<<16 = 512*512
    i32x4 g2; g2[0] = 4; g2[1] = 0; g2[2] = 0; g2[3] = 0;  // tensor_dim2 = 4
    i32x4 g3 = {0, 0, 0, 0};
    i32x8 g4 = {0, 0, 0, 0, 0, 0, 0, 0};          // extended group (unused)
    __builtin_amdgcn_tensor_load_to_lds(g0, g1, g2, g3, g4, 0);
    __builtin_amdgcn_s_wait_tensorcnt(0);
  }
  __syncthreads();
#else
  // Fallback: per-lane async->LDS gathers of the 32 owned gate rows.
  for (int idx = tid; idx < (32 * HID) / 8; idx += 256) {
    int lr   = idx >> 6;            // 64 16B-chunks per row
    int col  = (idx & 63) << 3;
    int grow = ((lr >> 3) * HID) + j0 + (lr & 7);
    async_load_b128(wlds + lr * HID + col, Whh + (size_t)grow * HID + col);
  }
  wait_async0();
  __syncthreads();
#endif

  const int r   = tid >> 3;         // local gate row 0..31
  const int seg = tid & 7;          // 64-wide K segment
  float c_state = 0.f;              // valid for tid < 8

  for (int t = 0; t < nsteps; ++t) {
    if (t == 0) {
      if (tid < 128) *(unsigned*)(hlds + tid * 4) = 0u;  // zero 2 bf16/thread at stride 4 (512B of the 1KB)
    } else {
      if (tid < 64)  // 64 chunks of 16B = 512 bf16
        async_load_b128(hlds + tid * 8, Hout + (size_t)(t - 1) * HID + tid * 8);
      wait_async0();
    }
    __syncthreads();

    const unsigned short* wrow = wlds + r * HID + seg * 64;
    const unsigned short* hseg = hlds + seg * 64;
    float s = 0.f;
    #pragma unroll
    for (int k = 0; k < 64; ++k) s += bf2f(wrow[k]) * bf2f(hseg[k]);
    red[r][seg] = s;
    __syncthreads();

    if (tid < 32) {
      float d = 0.f;
      #pragma unroll
      for (int q = 0; q < 8; ++q) d += red[tid][q];
      int grow = ((tid >> 3) * HID) + j0 + (tid & 7);
      d += G[(size_t)t * GATES + grow];
      red[tid][0] = d;
    }
    __syncthreads();

    if (tid < 8) {
      float iv = 1.f / (1.f + __expf(-red[tid][0]));
      float fv = 1.f / (1.f + __expf(-red[8 + tid][0]));
      float gv = tanhf(red[16 + tid][0]);
      float ov = 1.f / (1.f + __expf(-red[24 + tid][0]));
      c_state  = fv * c_state + iv * gv;
      float h  = ov * tanhf(c_state);
      Hout[(size_t)t * HID + j0 + tid] = f2bf(h);
    }
    __threadfence();
    __syncthreads();
    if (tid == 0) {                 // grid-wide generation barrier
      int gen = atomicAdd(bar + 1, 0);
      if (atomicAdd(bar, 1) == (int)gridDim.x - 1) {
        atomicExch(bar, 0);
        atomicAdd(bar + 1, 1);
      } else {
        while (atomicAdd(bar + 1, 0) == gen) __builtin_amdgcn_s_sleep(2);
      }
    }
    __syncthreads();
  }
}

// ---------------- host orchestration ----------------
extern "C" void kernel_launch(void* const* d_in, const int* in_sizes, int n_in,
                              void* d_out, int out_size, void* d_ws, size_t ws_size,
                              hipStream_t stream)
{
  (void)in_sizes; (void)n_in; (void)out_size; (void)ws_size;

  const float* x = (const float*)d_in[0];
  const float *Wih[4], *Whh[4], *bih[4], *bhh[4];
  for (int l = 0; l < 4; ++l) {
    Wih[l] = (const float*)d_in[1 + 4 * l];
    Whh[l] = (const float*)d_in[2 + 4 * l];
    bih[l] = (const float*)d_in[3 + 4 * l];
    bhh[l] = (const float*)d_in[4 + 4 * l];
  }
  const float* Wout = (const float*)d_in[17];
  const float* bout = (const float*)d_in[18];

  // Workspace layout (bytes), ~57.7 MB total.
  char* ws = (char*)d_ws;
  float*          G     = (float*)(ws + 0);                  // 4096*2048*4 = 32MB
  unsigned short* X0    = (unsigned short*)(ws + 33554432);  // 4096*1024*2 = 8MB
  unsigned short* XA    = (unsigned short*)(ws + 41943040);  // 4096*512*2  = 4MB
  unsigned short* XB    = (unsigned short*)(ws + 46137344);  // 4MB
  unsigned short* WihB  = (unsigned short*)(ws + 50331648);  // 2048*1024*2 = 4MB
  unsigned short* WhhB  = (unsigned short*)(ws + 54525952);  // 2048*512*2  = 2MB
  unsigned short* WoutB = (unsigned short*)(ws + 56623104);  // 1024*512*2  = 1MB
  float*          biasC = (float*)(ws + 57671680);           // 2048*4
  int*            bar   = (int*)(ws + 57679872);             // barrier state

  init_barrier_kernel<<<1, 32, 0, stream>>>(bar);

  { // input -> bf16
    int n = T_STEPS * 1024;
    cvt_f32_bf16_kernel<<<(n + 255) / 256, 256, 0, stream>>>(x, X0, n);
  }

  const unsigned short* Xin = X0;
  unsigned short* bufs[2] = { XA, XB };
  for (int l = 0; l < 4; ++l) {
    int K  = (l == 0) ? 1024 : 512;
    int nW = GATES * K;
    cvt_f32_bf16_kernel<<<(nW + 255) / 256, 256, 0, stream>>>(Wih[l], WihB, nW);
    int nH = GATES * HID;
    cvt_f32_bf16_kernel<<<(nH + 255) / 256, 256, 0, stream>>>(Whh[l], WhhB, nH);
    add_vec_kernel<<<(GATES + 255) / 256, 256, 0, stream>>>(bih[l], bhh[l], biasC, GATES);

    dim3 grid(GATES / TN, T_STEPS / TM);
    gemm_bias_wmma_kernel<<<grid, 256, 0, stream>>>(Xin, WihB, biasC, G,
                                                    T_STEPS, GATES, K);

    unsigned short* Hout = bufs[l & 1];
    lstm_scan_kernel<<<64, 256, 0, stream>>>(G, WhhB, Hout, bar, T_STEPS);
    Xin = Hout;
  }

  { // output projection -> d_out (fp32)
    int n = 1024 * HID;
    cvt_f32_bf16_kernel<<<(n + 255) / 256, 256, 0, stream>>>(Wout, WoutB, n);
    dim3 grid(1024 / TN, T_STEPS / TM);
    gemm_bias_wmma_kernel<<<grid, 256, 0, stream>>>(Xin, WoutB, bout,
                                                    (float*)d_out, T_STEPS, 1024, HID);
  }
}
